// SpGraphAttentionLayer_81827716924046
// MI455X (gfx1250) — matrix operand order repaired and checked by reading, not compiled, not run on hardware
//
#include <hip/hip_runtime.h>
#include <hip/hip_bf16.h>

// Problem constants (match reference: B=8, N=2500, IN_FEAT=128, NHID=256, E=320000)
#define NUM_NODES 20000
#define NE        320000
#define FDIM      128
#define NHIDC     256
#define KATT      288   // 2*128+3 padded up to 9*32
#define K1P       160   // 128+3 padded up to 5*32

typedef __attribute__((ext_vector_type(8)))  _Float16 v8h;
typedef __attribute__((ext_vector_type(16))) _Float16 v16h;
typedef __attribute__((ext_vector_type(8)))  float    v8f;

union V16U { v16h v; v8h h[2]; };

// ---------------------------------------------------------------------------
// WMMA fragment loaders (wave32, per CDNA5 ISA 7.12.2 layouts)
// A (16x32 f16, row-major source, stride in halves):
//   lanes 0-15: row=lane, halves 0..7 = K 0..7, halves 8..15 = K 16..23
//   lanes 16-31: row=lane-16, K 8..15 then 24..31
__device__ __forceinline__ v16h load_a_frag(const _Float16* base, int strideK,
                                            int k0, int lane) {
  int row  = lane & 15;
  int koff = k0 + ((lane & 16) ? 8 : 0);
  const _Float16* p = base + row * strideK + koff;
  V16U u;
  u.h[0] = *(const v8h*)(p);
  u.h[1] = *(const v8h*)(p + 16);
  return u.v;
}

// B (32x16 f16), weights stored K-contiguous per output column: Wt[n][k]
//   lanes 0-15: col=lane, K 0..7 then 16..23 ; lanes 16-31: col=lane-16, K 8..15 then 24..31
__device__ __forceinline__ v16h load_b_frag(const _Float16* __restrict__ Wt,
                                            int strideK, int k0, int ncol, int lane) {
  int col  = ncol + (lane & 15);
  int koff = k0 + ((lane & 16) ? 8 : 0);
  const _Float16* p = Wt + (size_t)col * strideK + koff;
  V16U u;
  u.h[0] = *(const v8h*)(p);
  u.h[1] = *(const v8h*)(p + 16);
  return u.v;
}

__device__ __forceinline__ v8f wmma16(v16h a, v16h b, v8f c) {
  return __builtin_amdgcn_wmma_f32_16x16x32_f16(false, a, false, b,
                                                (short)0, c, false, false);
}

// ---------------------------------------------------------------------------
// Pre-pass: f32 -> f16 node features
__global__ void k_conv_nodes(const float* __restrict__ Xk, const float* __restrict__ Xv,
                             _Float16* __restrict__ hk, _Float16* __restrict__ hv, int n) {
  int i = blockIdx.x * 256 + threadIdx.x;
  if (i < n) { hk[i] = (_Float16)Xk[i]; hv[i] = (_Float16)Xv[i]; }
}

// Pre-pass: W (Kreal x NHID, row-major f32) -> Wt (NHID x Kpad, K-contiguous f16, zero pad)
__global__ void k_build_wt(const float* __restrict__ W, _Float16* __restrict__ Wt,
                           int Kreal, int Kpad) {
  int i = blockIdx.x * 256 + threadIdx.x;
  if (i >= NHIDC * Kpad) return;
  int n = i / Kpad, k = i - n * Kpad;
  Wt[i] = (k < Kreal) ? (_Float16)W[k * NHIDC + n] : (_Float16)0.f;
}

__global__ void k_init(float* __restrict__ att_max, float* __restrict__ denom,
                       float* __restrict__ h_att_p) {
  int i = blockIdx.x * 256 + threadIdx.x;
  if (i < NUM_NODES) { att_max[i] = -INFINITY; denom[i] = 0.f; }
  if (i < NUM_NODES * K1P) h_att_p[i] = 0.f;
}

// ---------------------------------------------------------------------------
// Edge attention scores: per block = 32 edges. A = gathered [hv[src], hk[dst], ef]
// (32 x KATT f16) in LDS; each wave computes a 2x2 tile (2 M-tiles x 2 N-tiles),
// reusing each B fragment for two WMMAs (halves W_att cache traffic).
// Fused epilogue: tanh(+b_att) dot va -> att_sim[e].
__global__ __launch_bounds__(256) void k_edge_att(
    const _Float16* __restrict__ hv, const _Float16* __restrict__ hk,
    const float* __restrict__ ef, const int* __restrict__ esrc,
    const int* __restrict__ edst, const _Float16* __restrict__ WtA,
    const float* __restrict__ b_att, const float* __restrict__ va,
    float* __restrict__ att_sim) {
  __shared__ __align__(32) _Float16 As[32 * KATT];   // 18 KB
  __shared__ float red[32];

  int t = threadIdx.x;
  int lane = t & 31, wave = t >> 5;
  int e0 = blockIdx.x * 32;
  if (t < 32) red[t] = 0.f;

  // Gather A tile: 8 threads per edge row, 2x v8h (16B) loads per 128-wide segment
  int row = t >> 3, tc = t & 7;
  int e  = e0 + row;
  int ec = (e < NE) ? e : (NE - 1);
  int src = esrc[ec], dst = edst[ec];
  {
    const _Float16* pv = hv + (size_t)src * FDIM + tc * 16;
    const _Float16* pk = hk + (size_t)dst * FDIM + tc * 16;
    _Float16* dstv = &As[row * KATT + tc * 16];
    _Float16* dstk = &As[row * KATT + FDIM + tc * 16];
    *(v8h*)(dstv)     = *(const v8h*)(pv);
    *(v8h*)(dstv + 8) = *(const v8h*)(pv + 8);
    *(v8h*)(dstk)     = *(const v8h*)(pk);
    *(v8h*)(dstk + 8) = *(const v8h*)(pk + 8);
  }
  if (tc < 4) {
    v8h tmp = {};
    if (tc == 0) {
      tmp[0] = (_Float16)ef[ec * 3 + 0];
      tmp[1] = (_Float16)ef[ec * 3 + 1];
      tmp[2] = (_Float16)ef[ec * 3 + 2];
    }
    *(v8h*)&As[row * KATT + 2 * FDIM + tc * 8] = tmp;
  }
  __syncthreads();

  // 8 waves x (2 M-tiles x 2 N-tiles); K loop over 9 chunks of 32
  v8f acc00 = {}, acc01 = {}, acc10 = {}, acc11 = {};
  int n0 = wave * 32;
#pragma unroll
  for (int kc = 0; kc < KATT / 32; ++kc) {
    v16h b0 = load_b_frag(WtA, KATT, kc * 32, n0, lane);
    v16h b1 = load_b_frag(WtA, KATT, kc * 32, n0 + 16, lane);
    v16h a0 = load_a_frag(As, KATT, kc * 32, lane);
    v16h a1 = load_a_frag(As + 16 * KATT, KATT, kc * 32, lane);
    acc00 = wmma16(a0, b0, acc00);
    acc01 = wmma16(a0, b1, acc01);
    acc10 = wmma16(a1, b0, acc10);
    acc11 = wmma16(a1, b1, acc11);
  }

  // Epilogue: s = tanh(acc + b_att[col]) * va[col]; reduce over 256 cols per row
  int col = n0 + (lane & 15);
  int rb  = (lane & 16) ? 8 : 0;
  float ba0 = b_att[col], ba1 = b_att[col + 16];
  float va0 = va[col],    va1 = va[col + 16];
#pragma unroll
  for (int v = 0; v < 8; ++v) {
    float p = tanhf(acc00[v] + ba0) * va0 + tanhf(acc01[v] + ba1) * va1;
    p += __shfl_xor(p, 1, 32);
    p += __shfl_xor(p, 2, 32);
    p += __shfl_xor(p, 4, 32);
    p += __shfl_xor(p, 8, 32);
    if ((lane & 15) == 0) atomicAdd(&red[rb + v], p);
  }
#pragma unroll
  for (int v = 0; v < 8; ++v) {
    float p = tanhf(acc10[v] + ba0) * va0 + tanhf(acc11[v] + ba1) * va1;
    p += __shfl_xor(p, 1, 32);
    p += __shfl_xor(p, 2, 32);
    p += __shfl_xor(p, 4, 32);
    p += __shfl_xor(p, 8, 32);
    if ((lane & 15) == 0) atomicAdd(&red[16 + rb + v], p);
  }
  __syncthreads();
  if (t < 32 && (e0 + t) < NE) att_sim[e0 + t] = red[t];
}

// ---------------------------------------------------------------------------
// Segment max via signed/unsigned atomic trick on float bits
__global__ void k_att_max(const float* __restrict__ att_sim, const int* __restrict__ edst,
                          float* __restrict__ att_max) {
  int i = blockIdx.x * 256 + threadIdx.x;
  if (i >= NE) return;
  float v = att_sim[i];
  int d = edst[i];
  if (v >= 0.f) atomicMax((int*)&att_max[d], __float_as_int(v));
  else          atomicMin((unsigned int*)&att_max[d], (unsigned int)__float_as_int(v));
}

__global__ void k_exp(const float* __restrict__ att_sim, const int* __restrict__ edst,
                      const float* __restrict__ att_max, float* __restrict__ e_val,
                      float* __restrict__ denom) {
  int i = blockIdx.x * 256 + threadIdx.x;
  if (i >= NE) return;
  int d = edst[i];
  float ev = expf(att_sim[i] - att_max[d]);
  e_val[i] = ev;
  atomicAdd(&denom[d], ev);
}

// Weighted scatter: h_att_p[dst] += alpha * [h_value[src](128), edge_feat(3)], one wave/edge
__global__ void k_agg(const float* __restrict__ e_val, const float* __restrict__ denom,
                      const int* __restrict__ esrc, const int* __restrict__ edst,
                      const float* __restrict__ Xv, const float* __restrict__ ef,
                      float* __restrict__ h_att_p) {
  int idx  = blockIdx.x * 256 + threadIdx.x;
  int edge = idx >> 5;
  int lane = idx & 31;
  if (edge >= NE) return;
  int d = edst[edge];
  float alpha = e_val[edge] / denom[d];
  const float* sh = Xv + (size_t)esrc[edge] * FDIM;
  float* out = h_att_p + (size_t)d * K1P;
#pragma unroll
  for (int c = lane; c < FDIM; c += 32) atomicAdd(&out[c], alpha * sh[c]);
  if (lane < 3) atomicAdd(&out[FDIM + lane], alpha * ef[edge * 3 + lane]);
}

// ---------------------------------------------------------------------------
// Fused MLP: relu(relu(h_att @ W1 + b1) @ W2 + b2), 16 nodes per block, all WMMA
__global__ __launch_bounds__(256) void k_mlp(
    const float* __restrict__ h_att_p, const _Float16* __restrict__ Wt1,
    const float* __restrict__ b1, const _Float16* __restrict__ Wt2,
    const float* __restrict__ b2, float* __restrict__ out) {
  __shared__ __align__(32) _Float16 A1[16 * K1P];
  __shared__ __align__(32) _Float16 H[16 * NHIDC];

  int t = threadIdx.x, lane = t & 31, wave = t >> 5;
  int row = t >> 4, tc = t & 15;
  int node  = blockIdx.x * 16 + row;
  int nodec = (node < NUM_NODES) ? node : (NUM_NODES - 1);
  const float* srcp = h_att_p + (size_t)nodec * K1P;
#pragma unroll
  for (int j = 0; j < K1P / 16; ++j) {
    int c = tc + j * 16;
    A1[row * K1P + c] = (_Float16)srcp[c];
  }
  __syncthreads();

  int n0 = wave * 32;
  v8f c0 = {}, c1 = {};
#pragma unroll
  for (int kc = 0; kc < K1P / 32; ++kc) {
    v16h a = load_a_frag(A1, K1P, kc * 32, lane);
    c0 = wmma16(a, load_b_frag(Wt1, K1P, kc * 32, n0, lane), c0);
    c1 = wmma16(a, load_b_frag(Wt1, K1P, kc * 32, n0 + 16, lane), c1);
  }
  int col = n0 + (lane & 15);
  int rb  = (lane & 16) ? 8 : 0;
  float bb0 = b1[col], bb1 = b1[col + 16];
#pragma unroll
  for (int v = 0; v < 8; ++v) {
    H[(rb + v) * NHIDC + col]      = (_Float16)fmaxf(c0[v] + bb0, 0.f);
    H[(rb + v) * NHIDC + col + 16] = (_Float16)fmaxf(c1[v] + bb1, 0.f);
  }
  __syncthreads();

  v8f d0 = {}, d1 = {};
#pragma unroll
  for (int kc = 0; kc < NHIDC / 32; ++kc) {
    v16h a = load_a_frag(H, NHIDC, kc * 32, lane);
    d0 = wmma16(a, load_b_frag(Wt2, NHIDC, kc * 32, n0, lane), d0);
    d1 = wmma16(a, load_b_frag(Wt2, NHIDC, kc * 32, n0 + 16, lane), d1);
  }
  float cb0 = b2[col], cb1 = b2[col + 16];
  int nodeb = blockIdx.x * 16;
#pragma unroll
  for (int v = 0; v < 8; ++v) {
    int nn = nodeb + rb + v;
    if (nn < NUM_NODES) {
      out[(size_t)nn * NHIDC + col]      = fmaxf(d0[v] + cb0, 0.f);
      out[(size_t)nn * NHIDC + col + 16] = fmaxf(d1[v] + cb1, 0.f);
    }
  }
}

// ---------------------------------------------------------------------------
extern "C" void kernel_launch(void* const* d_in, const int* in_sizes, int n_in,
                              void* d_out, int out_size, void* d_ws, size_t ws_size,
                              hipStream_t stream) {
  const float* Xk    = (const float*)d_in[0];
  const float* Xv    = (const float*)d_in[1];
  const float* ef    = (const float*)d_in[2];
  const float* W_att = (const float*)d_in[3];
  const float* b_att = (const float*)d_in[4];
  const float* va    = (const float*)d_in[5];
  const float* W1    = (const float*)d_in[6];
  const float* b1    = (const float*)d_in[7];
  const float* W2    = (const float*)d_in[8];
  const float* b2    = (const float*)d_in[9];
  const int*   esrc  = (const int*)d_in[10];
  const int*   edst  = (const int*)d_in[11];
  float* out = (float*)d_out;

  char* ws = (char*)d_ws;
  size_t off = 0;
  auto alloc = [&](size_t bytes) -> void* {
    void* p = ws + off;
    off = (off + bytes + 255) & ~(size_t)255;
    return p;
  };
  _Float16* hv      = (_Float16*)alloc((size_t)NUM_NODES * FDIM * 2);
  _Float16* hk      = (_Float16*)alloc((size_t)NUM_NODES * FDIM * 2);
  _Float16* WtA     = (_Float16*)alloc((size_t)NHIDC * KATT * 2);
  _Float16* Wt1     = (_Float16*)alloc((size_t)NHIDC * K1P * 2);
  _Float16* Wt2     = (_Float16*)alloc((size_t)NHIDC * NHIDC * 2);
  float*    att_sim = (float*)alloc((size_t)NE * 4);
  float*    e_val   = (float*)alloc((size_t)NE * 4);
  float*    att_max = (float*)alloc((size_t)NUM_NODES * 4);
  float*    denom   = (float*)alloc((size_t)NUM_NODES * 4);
  float*    h_att_p = (float*)alloc((size_t)NUM_NODES * K1P * 4);

  k_conv_nodes<<<(NUM_NODES * FDIM + 255) / 256, 256, 0, stream>>>(Xk, Xv, hk, hv,
                                                                   NUM_NODES * FDIM);
  k_build_wt<<<(NHIDC * KATT + 255) / 256, 256, 0, stream>>>(W_att, WtA, 2 * FDIM + 3, KATT);
  k_build_wt<<<(NHIDC * K1P + 255) / 256, 256, 0, stream>>>(W1, Wt1, FDIM + 3, K1P);
  k_build_wt<<<(NHIDC * NHIDC + 255) / 256, 256, 0, stream>>>(W2, Wt2, NHIDC, NHIDC);
  k_init<<<(NUM_NODES * K1P + 255) / 256, 256, 0, stream>>>(att_max, denom, h_att_p);

  k_edge_att<<<(NE + 31) / 32, 256, 0, stream>>>(hv, hk, ef, esrc, edst, WtA, b_att, va,
                                                 att_sim);
  k_att_max<<<(NE + 255) / 256, 256, 0, stream>>>(att_sim, edst, att_max);
  k_exp<<<(NE + 255) / 256, 256, 0, stream>>>(att_sim, edst, att_max, e_val, denom);
  k_agg<<<((size_t)NE * 32 + 255) / 256, 256, 0, stream>>>(e_val, denom, esrc, edst, Xv, ef,
                                                           h_att_p);
  k_mlp<<<(NUM_NODES + 15) / 16, 256, 0, stream>>>(h_att_p, Wt1, b1, Wt2, b2, out);
}
